// MultiScaleRetention_68796786147434
// MI455X (gfx1250) — matrix operand order repaired and checked by reading, not compiled, not run on hardware
//
#include <hip/hip_runtime.h>
#include <hip/hip_bf16.h>

// ---------------------------------------------------------------------------
// MultiScaleRetention for MI455X (gfx1250): bf16 WMMA + async global->LDS
// ---------------------------------------------------------------------------

#define S_LEN 2048
#define D_DIM 1024
#define NH    16
#define KDIM  64
#define VDIM  2048
#define HSZ   128

typedef __attribute__((ext_vector_type(16))) __bf16       bf16x16;
typedef __attribute__((ext_vector_type(8)))  float        f32x8;
typedef __attribute__((ext_vector_type(4)))  unsigned int u32x4;

union FragU { u32x4 q[2]; bf16x16 v; };

static __device__ __forceinline__ unsigned short f32_to_bf16_bits(float f) {
  unsigned int u = __float_as_uint(f);
  u += 0x7FFFu + ((u >> 16) & 1u);   // round-to-nearest-even
  return (unsigned short)(u >> 16);
}

// Async DMA: global -> LDS, 16B per lane, tracked by ASYNCcnt.
// LDS byte address = low 32 bits of the generic pointer (ISA 10.2 aperture).
static __device__ __forceinline__ void async_copy_b128(void* lds_dst,
                                                       const void* gsrc) {
  asm volatile("global_load_async_to_lds_b128 %0, %1, off"
               :: "v"((unsigned int)(unsigned long long)lds_dst), "v"(gsrc)
               : "memory");
}
static __device__ __forceinline__ void wait_async_zero() {
  asm volatile("s_wait_asynccnt 0" ::: "memory");
}

// A fragment: 16x32 bf16, row-major source with leading dim ld (elements).
// lanes 0-15: row=lane,   e0..7 = K0..7,  e8..15 = K16..23
// lanes16-31: row=lane-16,e0..7 = K8..15, e8..15 = K24..31
static __device__ __forceinline__ bf16x16
load_frag_a(const unsigned short* base, int row, int ld) {
  int lane = threadIdx.x & 31;
  const unsigned short* p = base + (size_t)(row + (lane & 15)) * ld;
  int kh = (lane >> 4) * 8;
  FragU f;
  f.q[0] = *(const u32x4*)(p + kh);
  f.q[1] = *(const u32x4*)(p + 16 + kh);
  return f.v;
}

// B fragment: 32x16 bf16 supplied as Bt[N][K] row-major with leading dim ld.
// lanes 0-15: col=lane, e0..7 = K0..7, e8..15 = K16..23 ; lanes 16-31: K8..15, K24..31
static __device__ __forceinline__ bf16x16
load_frag_b(const unsigned short* base, int col, int ld) {
  int lane = threadIdx.x & 31;
  const unsigned short* p = base + (size_t)(col + (lane & 15)) * ld;
  int kh = (lane >> 4) * 8;
  FragU f;
  f.q[0] = *(const u32x4*)(p + kh);
  f.q[1] = *(const u32x4*)(p + 16 + kh);
  return f.v;
}

static __device__ __forceinline__ f32x8
wmma_bf16(bf16x16 a, bf16x16 b, f32x8 c) {
  return __builtin_amdgcn_wmma_f32_16x16x32_bf16(
      /*neg_a=*/false, a, /*neg_b=*/false, b,
      /*c_mod=*/(short)0, c, /*reuse_a=*/false, /*reuse_b=*/false);
}

// ---------------------------------------------------------------------------
// f32 -> bf16 flat convert
// ---------------------------------------------------------------------------
__global__ void msr_cvt_bf16(const float* __restrict__ in,
                             unsigned short* __restrict__ out, int n) {
  int i = blockIdx.x * blockDim.x + threadIdx.x;
  if (i < n) out[i] = f32_to_bf16_bits(in[i]);
}

// ---------------------------------------------------------------------------
// f32 [R][C] -> bf16 [C][R] (transpose + convert), 32x32 LDS tiles
// ---------------------------------------------------------------------------
__global__ __launch_bounds__(256) void msr_transpose_bf16(
    const float* __restrict__ V, unsigned short* __restrict__ Vt,
    int R, int C) {
  __shared__ unsigned short tile[32][33];
  int bx = blockIdx.x * 32;  // col base
  int by = blockIdx.y * 32;  // row base
  int tx = threadIdx.x & 31, ty = threadIdx.x >> 5;
  for (int r = ty; r < 32; r += 8)
    tile[r][tx] = f32_to_bf16_bits(V[(size_t)(by + r) * C + bx + tx]);
  __syncthreads();
  for (int r = ty; r < 32; r += 8)
    Vt[(size_t)(bx + r) * R + by + tx] = tile[tx][r];
}

// ---------------------------------------------------------------------------
// Generic bf16 GEMM: C[M][N] f32 = A[M][K] @ Bt[N][K]^T
// 128x128 tile per workgroup, 8 waves (4x2), each wave 32x64 of C.
// Double-buffered async global->LDS staging.
// ---------------------------------------------------------------------------
__global__ __launch_bounds__(256) void msr_gemm_bf16(
    const unsigned short* __restrict__ A,
    const unsigned short* __restrict__ Bt,
    float* __restrict__ C, int M, int N, int K) {
  extern __shared__ char smem[];
  unsigned short* As0 = (unsigned short*)smem;   // 128*32
  unsigned short* Bs0 = As0 + 128 * 32;
  unsigned short* As1 = Bs0 + 128 * 32;
  unsigned short* Bs1 = As1 + 128 * 32;

  int tid = threadIdx.x, lane = tid & 31, wave = tid >> 5;
  int wm = wave & 3, wn = wave >> 2;
  int bm = blockIdx.y * 128, bn = blockIdx.x * 128;

  auto stage = [&](unsigned short* as, unsigned short* bs, int kb) {
    for (int it = tid; it < 512; it += 256) {
      int row = it >> 2, c4 = (it & 3) * 8;
      async_copy_b128(as + row * 32 + c4, A + (size_t)(bm + row) * K + kb + c4);
      async_copy_b128(bs + row * 32 + c4, Bt + (size_t)(bn + row) * K + kb + c4);
    }
  };

  f32x8 acc[2][4] = {};
  int nk = K >> 5;
  stage(As0, Bs0, 0);

  for (int ki = 0; ki < nk; ++ki) {
    wait_async_zero();
    __syncthreads();
    unsigned short* as = (ki & 1) ? As1 : As0;
    unsigned short* bs = (ki & 1) ? Bs1 : Bs0;
    if (ki + 1 < nk)
      stage((ki & 1) ? As0 : As1, (ki & 1) ? Bs0 : Bs1, (ki + 1) * 32);

    bf16x16 a0 = load_frag_a(as, wm * 32, 32);
    bf16x16 a1 = load_frag_a(as, wm * 32 + 16, 32);
#pragma unroll
    for (int j = 0; j < 4; ++j) {
      bf16x16 bj = load_frag_b(bs, wn * 64 + j * 16, 32);
      acc[0][j] = wmma_bf16(a0, bj, acc[0][j]);
      acc[1][j] = wmma_bf16(a1, bj, acc[1][j]);
    }
    __syncthreads();
  }

  int crow = (lane >> 4) * 8, ccol = lane & 15;
#pragma unroll
  for (int i = 0; i < 2; ++i)
#pragma unroll
    for (int j = 0; j < 4; ++j)
#pragma unroll
      for (int r = 0; r < 8; ++r)
        C[(size_t)(bm + wm * 32 + i * 16 + crow + r) * N +
          bn + wn * 64 + j * 16 + ccol] = acc[i][j][r];
}

// ---------------------------------------------------------------------------
// Rotary on q/k raw projections; fold k scaling; repack to [H][S][64] bf16
// ---------------------------------------------------------------------------
__global__ void msr_rotary(const float* __restrict__ qraw,
                           const float* __restrict__ kraw,
                           const float* __restrict__ sinp,
                           const float* __restrict__ cosp,
                           unsigned short* __restrict__ qbf,
                           unsigned short* __restrict__ kbf) {
  int idx = blockIdx.x * blockDim.x + threadIdx.x;  // pair index
  if (idx >= S_LEN * D_DIM / 2) return;
  int d2 = idx % (D_DIM / 2);
  int s  = idx / (D_DIM / 2);
  int d  = d2 * 2;
  int h  = d >> 6;       // / KDIM
  int dk = d & 63;

  float c0 = cosp[s * KDIM + dk], c1 = cosp[s * KDIM + dk + 1];
  float s0 = sinp[s * KDIM + dk], s1 = sinp[s * KDIM + dk + 1];

  float q0 = qraw[(size_t)s * D_DIM + d], q1 = qraw[(size_t)s * D_DIM + d + 1];
  float k0 = kraw[(size_t)s * D_DIM + d], k1 = kraw[(size_t)s * D_DIM + d + 1];

  // rotate_every_two: rot[2i] = -x[2i+1], rot[2i+1] = x[2i]
  float qr0 = q0 * c0 - q1 * s0;
  float qr1 = q1 * c1 + q0 * s1;
  const float scale = 0.125f;  // KD^-0.5
  float kr0 = (k0 * c0 - k1 * s0) * scale;
  float kr1 = (k1 * c1 + k0 * s1) * scale;

  size_t o = ((size_t)h * S_LEN + s) * KDIM + dk;
  qbf[o]     = f32_to_bf16_bits(qr0);
  qbf[o + 1] = f32_to_bf16_bits(qr1);
  kbf[o]     = f32_to_bf16_bits(kr0);
  kbf[o + 1] = f32_to_bf16_bits(kr1);
}

// ---------------------------------------------------------------------------
// Fused retention: per (head, 128-row block)
//   pass 1: den_s = |sum_t (q@k^T)*mask| clipped to >= 1
//   pass 2: o = ((q@k^T)*mask/den) @ v, then group-norm + SiLU(g) gate
// qbf/kbf: [H][S][64] bf16 ; vtbf: [H][128][S] bf16
// k/v tiles double-buffered via async global->LDS DMA.
// ---------------------------------------------------------------------------
__global__ __launch_bounds__(256) void msr_retention(
    const unsigned short* __restrict__ qbf,
    const unsigned short* __restrict__ kbf,
    const unsigned short* __restrict__ vtbf,
    const float* __restrict__ mask,
    const float* __restrict__ g,
    const float* __restrict__ gn_w,
    const float* __restrict__ gn_b,
    unsigned short* __restrict__ att) {
  extern __shared__ char smem[];
  unsigned short* qs   = (unsigned short*)smem;   // 128*64
  unsigned short* ks0  = qs + 128 * 64;           // 128*64
  unsigned short* ks1  = ks0 + 128 * 64;          // 128*64
  unsigned short* qkbf = ks1 + 128 * 64;          // 128*128
  unsigned short* vs0  = qkbf + 128 * 128;        // 128*128
  unsigned short* vs1  = vs0 + 128 * 128;         // 128*128
  float* denb = (float*)(vs1 + 128 * 128);        // 128 f32
  float* obuf = (float*)smem;                     // 128*128 f32 alias (epilogue)

  int tid = threadIdx.x, lane = tid & 31, wave = tid >> 5;
  int wm = wave & 3, wn = wave >> 2;
  int crow = (lane >> 4) * 8, ccol = lane & 15;
  int h  = blockIdx.y;
  int sb = blockIdx.x * 128;

  auto stage_k = [&](unsigned short* kbuf, int tb) {
    const unsigned short* ksrc = kbf + ((size_t)h * S_LEN + tb) * KDIM;
    for (int it = tid; it < 1024; it += 256) {
      int row = it >> 3, c = (it & 7) * 8;
      async_copy_b128(kbuf + row * 64 + c, ksrc + (size_t)row * KDIM + c);
    }
  };
  auto stage_v = [&](unsigned short* vbuf, int tb) {
    const unsigned short* vsrc = vtbf + (size_t)h * HSZ * S_LEN + tb;
    for (int it = tid; it < 2048; it += 256) {
      int row = it >> 4, c = (it & 15) * 8;
      async_copy_b128(vbuf + row * 128 + c, vsrc + (size_t)row * S_LEN + c);
    }
  };

  // stage q block once (async as well): [128][64]
  {
    const unsigned short* qsrc = qbf + ((size_t)h * S_LEN + sb) * KDIM;
    for (int it = tid; it < 1024; it += 256) {
      int row = it >> 3, c = (it & 7) * 8;
      async_copy_b128(qs + row * 64 + c, qsrc + (size_t)row * KDIM + c);
    }
  }
  if (tid < 128) denb[tid] = 0.f;

  // ---------------- pass 1: row sums of masked qk ----------------
  float rs[2][8];
#pragma unroll
  for (int i = 0; i < 2; ++i)
#pragma unroll
    for (int r = 0; r < 8; ++r) rs[i][r] = 0.f;

  stage_k(ks0, 0);
  for (int tbi = 0; tbi < S_LEN / 128; ++tbi) {
    int tb = tbi * 128;
    wait_async_zero();
    __syncthreads();
    unsigned short* ks = (tbi & 1) ? ks1 : ks0;
    if (tbi + 1 < S_LEN / 128) stage_k((tbi & 1) ? ks0 : ks1, tb + 128);

    f32x8 acc[2][4] = {};
#pragma unroll
    for (int kk = 0; kk < 64; kk += 32) {
      bf16x16 a0 = load_frag_a(qs + kk, wm * 32, 64);
      bf16x16 a1 = load_frag_a(qs + kk, wm * 32 + 16, 64);
#pragma unroll
      for (int j = 0; j < 4; ++j) {
        bf16x16 bj = load_frag_b(ks + kk, wn * 64 + j * 16, 64);
        acc[0][j] = wmma_bf16(a0, bj, acc[0][j]);
        acc[1][j] = wmma_bf16(a1, bj, acc[1][j]);
      }
    }
#pragma unroll
    for (int i = 0; i < 2; ++i)
#pragma unroll
      for (int j = 0; j < 4; ++j) {
        int s_loc = wm * 32 + i * 16 + crow;
        int t_glb = tb + wn * 64 + j * 16 + ccol;
        const float* mrow =
            mask + ((size_t)h * S_LEN + sb + s_loc) * S_LEN + t_glb;
#pragma unroll
        for (int r = 0; r < 8; ++r)
          rs[i][r] += acc[i][j][r] *
                      __builtin_nontemporal_load(mrow + (size_t)r * S_LEN);
      }
    __syncthreads();
  }

  // reduce across 16 lanes sharing a row-set, then across wn waves
#pragma unroll
  for (int i = 0; i < 2; ++i)
#pragma unroll
    for (int r = 0; r < 8; ++r) {
      float v = rs[i][r];
      v += __shfl_xor(v, 1);
      v += __shfl_xor(v, 2);
      v += __shfl_xor(v, 4);
      v += __shfl_xor(v, 8);
      if (ccol == 0) atomicAdd(&denb[wm * 32 + i * 16 + crow], v);
    }
  __syncthreads();
  if (tid < 128) {
    float dsum = denb[tid];
    denb[tid] = 1.f / fmaxf(fabsf(dsum), 1.f);  // inv den
  }
  __syncthreads();

  // ---------------- pass 2: o = (qk*mask*inv_den) @ v ----------------
  f32x8 oacc[2][4] = {};

  stage_k(ks0, 0);
  stage_v(vs0, 0);
  for (int tbi = 0; tbi < S_LEN / 128; ++tbi) {
    int tb = tbi * 128;
    wait_async_zero();
    __syncthreads();
    unsigned short* ks = (tbi & 1) ? ks1 : ks0;
    unsigned short* vs = (tbi & 1) ? vs1 : vs0;
    if (tbi + 1 < S_LEN / 128) {
      stage_k((tbi & 1) ? ks0 : ks1, tb + 128);
      stage_v((tbi & 1) ? vs0 : vs1, tb + 128);
    }

    f32x8 acc[2][4] = {};
#pragma unroll
    for (int kk = 0; kk < 64; kk += 32) {
      bf16x16 a0 = load_frag_a(qs + kk, wm * 32, 64);
      bf16x16 a1 = load_frag_a(qs + kk, wm * 32 + 16, 64);
#pragma unroll
      for (int j = 0; j < 4; ++j) {
        bf16x16 bj = load_frag_b(ks + kk, wn * 64 + j * 16, 64);
        acc[0][j] = wmma_bf16(a0, bj, acc[0][j]);
        acc[1][j] = wmma_bf16(a1, bj, acc[1][j]);
      }
    }
    // scale, mask, convert; re-stripe through LDS (C-layout -> A-layout)
#pragma unroll
    for (int i = 0; i < 2; ++i)
#pragma unroll
      for (int j = 0; j < 4; ++j) {
        int s_loc = wm * 32 + i * 16 + crow;
        int t_loc = wn * 64 + j * 16 + ccol;
        const float* mrow =
            mask + ((size_t)h * S_LEN + sb + s_loc) * S_LEN + tb + t_loc;
#pragma unroll
        for (int r = 0; r < 8; ++r) {
          float val = acc[i][j][r] *
                      __builtin_nontemporal_load(mrow + (size_t)r * S_LEN) *
                      denb[s_loc + r];
          qkbf[(s_loc + r) * 128 + t_loc] = f32_to_bf16_bits(val);
        }
      }
    __syncthreads();

#pragma unroll
    for (int kk = 0; kk < 128; kk += 32) {
      bf16x16 a0 = load_frag_a(qkbf + kk, wm * 32, 128);
      bf16x16 a1 = load_frag_a(qkbf + kk, wm * 32 + 16, 128);
#pragma unroll
      for (int j = 0; j < 4; ++j) {
        bf16x16 bj = load_frag_b(vs + kk, wn * 64 + j * 16, 128);
        oacc[0][j] = wmma_bf16(a0, bj, oacc[0][j]);
        oacc[1][j] = wmma_bf16(a1, bj, oacc[1][j]);
      }
    }
    __syncthreads();
  }

  // ---------------- epilogue: group norm + SiLU(g) gate ----------------
#pragma unroll
  for (int i = 0; i < 2; ++i)
#pragma unroll
    for (int j = 0; j < 4; ++j)
#pragma unroll
      for (int r = 0; r < 8; ++r)
        obuf[(wm * 32 + i * 16 + crow + r) * 128 + wn * 64 + j * 16 + ccol] =
            oacc[i][j][r];
  __syncthreads();

  int row = tid >> 1, half = tid & 1;
  const float* orow = obuf + row * 128 + half * 64;
  float sum = 0.f, sq = 0.f;
#pragma unroll 8
  for (int c = 0; c < 64; ++c) {
    float v = orow[c];
    sum += v;
    sq += v * v;
  }
  sum += __shfl_xor(sum, 1);
  sq  += __shfl_xor(sq, 1);
  float mean = sum * (1.f / 128.f);
  float var  = sq * (1.f / 128.f) - mean * mean;
  float inv  = rsqrtf(var + 1e-5f);

  int s_glb = sb + row;
  int cbase = h * HSZ + half * 64;
#pragma unroll 8
  for (int c = 0; c < 64; ++c) {
    float v  = (orow[c] - mean) * inv;
    float w  = gn_w[cbase + c];
    float b  = gn_b[cbase + c];
    float gv = g[(size_t)s_glb * VDIM + cbase + c];
    float sig = 1.f / (1.f + __expf(-gv));
    att[(size_t)s_glb * VDIM + cbase + c] =
        f32_to_bf16_bits(gv * sig * (v * w + b));
  }
}

// ---------------------------------------------------------------------------
// Host launcher
// ---------------------------------------------------------------------------
static inline size_t align256(size_t x) { return (x + 255) & ~(size_t)255; }

extern "C" void kernel_launch(void* const* d_in, const int* in_sizes, int n_in,
                              void* d_out, int out_size, void* d_ws,
                              size_t ws_size, hipStream_t stream) {
  (void)in_sizes; (void)n_in; (void)out_size; (void)ws_size;
  const float* x    = (const float*)d_in[0];
  const float* sinp = (const float*)d_in[1];
  const float* cosp = (const float*)d_in[2];
  const float* mask = (const float*)d_in[3];
  const float* Wq   = (const float*)d_in[4];
  const float* Wk   = (const float*)d_in[5];
  const float* Wv   = (const float*)d_in[6];
  const float* Wg   = (const float*)d_in[7];
  const float* Wo   = (const float*)d_in[8];
  const float* gn_w = (const float*)d_in[9];
  const float* gn_b = (const float*)d_in[10];
  float* out = (float*)d_out;

  char* ws = (char*)d_ws;
  size_t off = 0;
  auto carve = [&](size_t bytes) {
    char* p = ws + off;
    off += align256(bytes);
    return p;
  };
  unsigned short* xb   = (unsigned short*)carve((size_t)S_LEN * D_DIM * 2);
  unsigned short* Wqt  = (unsigned short*)carve((size_t)D_DIM * D_DIM * 2);
  unsigned short* Wkt  = (unsigned short*)carve((size_t)D_DIM * D_DIM * 2);
  unsigned short* Wvt  = (unsigned short*)carve((size_t)D_DIM * VDIM * 2);
  unsigned short* Wgt  = (unsigned short*)carve((size_t)D_DIM * VDIM * 2);
  unsigned short* Wot  = (unsigned short*)carve((size_t)VDIM * D_DIM * 2);
  float* qraw = (float*)carve((size_t)S_LEN * D_DIM * 4);
  float* kraw = (float*)carve((size_t)S_LEN * D_DIM * 4);
  float* vraw = (float*)carve((size_t)S_LEN * VDIM * 4);
  float* graw = (float*)carve((size_t)S_LEN * VDIM * 4);
  unsigned short* qbf  = (unsigned short*)carve((size_t)NH * S_LEN * KDIM * 2);
  unsigned short* kbf  = (unsigned short*)carve((size_t)NH * S_LEN * KDIM * 2);
  unsigned short* vtbf = (unsigned short*)carve((size_t)VDIM * S_LEN * 2);
  unsigned short* attb = (unsigned short*)carve((size_t)S_LEN * VDIM * 2);

  // 1. convert x to bf16
  {
    int n = S_LEN * D_DIM;
    msr_cvt_bf16<<<(n + 255) / 256, 256, 0, stream>>>(x, xb, n);
  }
  // 2. convert + transpose weights: W[K][N] -> Wt[N][K] bf16
  msr_transpose_bf16<<<dim3(D_DIM / 32, D_DIM / 32), 256, 0, stream>>>(Wq, Wqt, D_DIM, D_DIM);
  msr_transpose_bf16<<<dim3(D_DIM / 32, D_DIM / 32), 256, 0, stream>>>(Wk, Wkt, D_DIM, D_DIM);
  msr_transpose_bf16<<<dim3(VDIM / 32, D_DIM / 32), 256, 0, stream>>>(Wv, Wvt, D_DIM, VDIM);
  msr_transpose_bf16<<<dim3(VDIM / 32, D_DIM / 32), 256, 0, stream>>>(Wg, Wgt, D_DIM, VDIM);
  msr_transpose_bf16<<<dim3(D_DIM / 32, VDIM / 32), 256, 0, stream>>>(Wo, Wot, VDIM, D_DIM);

  // 3. projections (bf16 WMMA GEMMs, double-buffered async staging)
  size_t gemm_lds = 4 * 128 * 32 * sizeof(unsigned short);  // 32 KB
  msr_gemm_bf16<<<dim3(D_DIM / 128, S_LEN / 128), 256, gemm_lds, stream>>>(
      xb, Wqt, qraw, S_LEN, D_DIM, D_DIM);
  msr_gemm_bf16<<<dim3(D_DIM / 128, S_LEN / 128), 256, gemm_lds, stream>>>(
      xb, Wkt, kraw, S_LEN, D_DIM, D_DIM);
  msr_gemm_bf16<<<dim3(VDIM / 128, S_LEN / 128), 256, gemm_lds, stream>>>(
      xb, Wvt, vraw, S_LEN, VDIM, D_DIM);
  msr_gemm_bf16<<<dim3(VDIM / 128, S_LEN / 128), 256, gemm_lds, stream>>>(
      xb, Wgt, graw, S_LEN, VDIM, D_DIM);

  // 4. rotary (+ k scaling), repack per head
  {
    int n = S_LEN * D_DIM / 2;
    msr_rotary<<<(n + 255) / 256, 256, 0, stream>>>(qraw, kraw, sinp, cosp,
                                                    qbf, kbf);
  }
  // 5. v transpose: vraw[S][VD] -> vtbf[VD][S] (== [H][128][S])
  msr_transpose_bf16<<<dim3(VDIM / 32, S_LEN / 32), 256, 0, stream>>>(
      vraw, vtbf, S_LEN, VDIM);

  // 6. fused retention + group norm + gate
  {
    size_t attn_lds =
        (size_t)(3 * 128 * 64 + 3 * 128 * 128) * sizeof(unsigned short) +
        128 * sizeof(float);  // ~145 KB
    msr_retention<<<dim3(S_LEN / 128, NH), 256, attn_lds, stream>>>(
        qbf, kbf, vtbf, mask, graw, gn_w, gn_b, attb);
  }

  // 7. final projection: out = att @ Wo
  msr_gemm_bf16<<<dim3(D_DIM / 128, S_LEN / 128), 256, gemm_lds, stream>>>(
      attb, Wot, out, S_LEN, D_DIM, VDIM);
}
